// MesoNet_40879498729139
// MI455X (gfx1250) — compile-verified, matrix operand order checked by reading, not run on hardware
//
#include <hip/hip_runtime.h>
#include <hip/hip_fp16.h>
#include <math.h>

typedef __attribute__((ext_vector_type(16))) _Float16 v16h;
typedef __attribute__((ext_vector_type(8)))  _Float16 v8h;
typedef __attribute__((ext_vector_type(8)))  float    v8f;

#define NN 40000
#define NE 100000

// workspace layout (float offsets)
#define WS_X2OUT 0
#define WS_AGG21 1280000
#define WS_AGG11 2560000
#define WS_CNT   3840000
#define WS_INTER 3880000
#define WS_GLOB  5160000
#define WS_GRP   6440000
#define WS_ATOM  7720000

// Build a v16h operand from two aligned 8-half runs (ISA 16-bit operand layout:
// lane grp=0/8 selects K runs [grp..grp+7] and [16+grp..16+grp+7]).
__device__ __forceinline__ v16h frag_from_runs(const _Float16* base, int grp) {
    v8h lo = *(const v8h*)(base + grp);
    v8h hi = *(const v8h*)(base + 16 + grp);
    return __builtin_shufflevector(lo, hi, 0, 1, 2, 3, 4, 5, 6, 7,
                                   8, 9, 10, 11, 12, 13, 14, 15);
}

__device__ __forceinline__ v8h cvt8(float4 a, float4 b) {
    v8h r;
    r[0] = (_Float16)a.x; r[1] = (_Float16)a.y; r[2] = (_Float16)a.z; r[3] = (_Float16)a.w;
    r[4] = (_Float16)b.x; r[5] = (_Float16)b.y; r[6] = (_Float16)b.z; r[7] = (_Float16)b.w;
    return r;
}

// ---------------------------------------------------------------- CfC + x22
__global__ void __launch_bounds__(256) cfc_kernel(
    const float* __restrict__ x,
    const float* __restrict__ ff1w, const float* __restrict__ ff1b,
    const float* __restrict__ ff2w, const float* __restrict__ ff2b,
    const float* __restrict__ taw,  const float* __restrict__ tab,
    const float* __restrict__ tbw,  const float* __restrict__ tbb,
    const float* __restrict__ x22w, const float* __restrict__ x22b,
    float* __restrict__ x2out)
{
    int n = blockIdx.x * blockDim.x + threadIdx.x;
    if (n >= NN) return;
    float x2r[6];
#pragma unroll
    for (int i = 0; i < 6; ++i) x2r[i] = x[(size_t)n * 115 + 41 + i];
    float h[12];
#pragma unroll
    for (int i = 0; i < 6; ++i) { h[i] = x2r[i]; h[6 + i] = x2r[i]; }
    float outs[30];
    for (int it = 0; it < 5; ++it) {
        float z[18];
#pragma unroll
        for (int i = 0; i < 6; ++i)  z[i] = x2r[i];
#pragma unroll
        for (int i = 0; i < 12; ++i) z[6 + i] = h[i];
        float f1[12], f2[12], tg[12];
        for (int o = 0; o < 12; ++o) {
            float a1 = ff1b[o], a2 = ff2b[o], at = tab[o] + tbb[o];
            for (int i = 0; i < 18; ++i) {
                float zi = z[i];
                a1 += zi * ff1w[i * 12 + o];
                a2 += zi * ff2w[i * 12 + o];
                at += zi * (taw[i * 12 + o] + tbw[i * 12 + o]);
            }
            f1[o] = tanhf(a1);
            f2[o] = tanhf(a2);
            tg[o] = 1.f / (1.f + __expf(-at));
        }
#pragma unroll
        for (int o = 0; o < 12; ++o) h[o] = f1[o] * (1.f - tg[o]) + tg[o] * f2[o];
#pragma unroll
        for (int i = 0; i < 6; ++i) outs[it * 6 + i] = h[i];
    }
    for (int c = 0; c < 32; ++c) {
        float a = x22b[c];
        for (int i = 0; i < 30; ++i) a += outs[i] * x22w[i * 32 + c];
        x2out[(size_t)n * 32 + c] = fmaxf(a, 0.f);
    }
}

// ---------------------------------------------------------------- edge count
__global__ void __launch_bounds__(256) count_kernel(const int* __restrict__ ei,
                                                    float* __restrict__ cnt)
{
    int e = blockIdx.x * blockDim.x + threadIdx.x;
    if (e >= NE) return;
    atomicAdd(&cnt[ei[NE + e]], 1.0f);
}

// ---------------------------------------------------------------- fused NNConv (WMMA)
// msg[e,o] = sum_p U[e,p]*B[p,o];  U[e, k*DIN+d] = h[e,k]*x_src[e,d];
// bias folded as K rows [32*DIN, 33*DIN): U = x_src[e,d], B = e2b row.
// B staged in LDS in WMMA-fragment order: sBf[(chunk*32+col)*32 + kk].
template <int DIN>
__global__ void __launch_bounds__(256) nnconv_kernel(
    const float* __restrict__ xin, int xstride,
    const float* __restrict__ e1w, const float* __restrict__ e1b,
    const float* __restrict__ e2w, const float* __restrict__ e2b,
    const int* __restrict__ ei, const float* __restrict__ eattr,
    float* __restrict__ agg)
{
    constexpr int KT     = 33 * DIN;
    constexpr int KPAD   = ((KT + 31) / 32) * 32;
    constexpr int CHUNKS = KPAD / 32;
    __shared__ __align__(16) _Float16 sBf[KPAD * 32];
    __shared__ float sh_h[8][16][32];
    __shared__ float sh_x[8][16][DIN];
    __shared__ int   sh_dst[8][16];
    __shared__ float s_e1w[320];
    __shared__ float s_e1b[32];

    const int tid  = threadIdx.x;
    const int wave = tid >> 5;
    const int lane = tid & 31;

    // stage B (f16, fragment order) and e1 params
    for (int idx = tid; idx < KPAD * 32; idx += 256) {
        int p = idx >> 5, o = idx & 31;
        float v;
        if (p < 32 * DIN)      v = e2w[(size_t)p * 32 + o];   // (k*DIN+d)*32+o == p*32+o
        else if (p < KT)       v = e2b[(p - 32 * DIN) * 32 + o];
        else                   v = 0.f;
        int cchunk = p >> 5, kk = p & 31;
        sBf[(cchunk << 10) + (o << 5) + kk] = (_Float16)v;
    }
    for (int idx = tid; idx < 320; idx += 256) s_e1w[idx] = e1w[idx];
    if (tid < 32) s_e1b[tid] = e1b[tid];
    __syncthreads();

    // per-wave 16-edge tile staging: h = relu(e1(edge_attr)), x_src row, dst
    const int m    = lane & 15;
    const int half = lane >> 4;
    const int grp  = half * 8;
    const int e    = (blockIdx.x * 8 + wave) * 16 + m;
    const bool valid = (e < NE);
    int srcN = valid ? ei[e] : 0;
    if (half == 0) sh_dst[wave][m] = valid ? ei[NE + e] : -1;
    float ea[10];
#pragma unroll
    for (int i = 0; i < 10; ++i) ea[i] = valid ? eattr[(size_t)e * 10 + i] : 0.f;
#pragma unroll
    for (int o16 = 0; o16 < 16; ++o16) {
        int o = half * 16 + o16;
        float a = s_e1b[o];
#pragma unroll
        for (int i = 0; i < 10; ++i) a += ea[i] * s_e1w[i * 32 + o];
        sh_h[wave][m][o] = valid ? fmaxf(a, 0.f) : 0.f;
    }
    for (int d = half; d < DIN; d += 2)
        sh_x[wave][m][d] = valid ? xin[(size_t)srcN * xstride + d] : 0.f;
    __syncthreads();

    // WMMA K-loop:  D(16x32) += A(16x32) * B(32x32)   (fully unrolled)
    v8f acc0 = {}, acc1 = {};
    const int ncol = lane & 15;
    const float* hr = sh_h[wave][m];
    const float* xr = sh_x[wave][m];
#pragma unroll
    for (int c = 0; c < CHUNKS; ++c) {
        v16h af;
#pragma unroll
        for (int run = 0; run < 2; ++run) {
#pragma unroll
            for (int t = 0; t < 8; ++t) {
                int p = c * 32 + run * 16 + grp + t;
                float av;
                if (p < 32 * DIN) {
                    int k = p / DIN;
                    int d = p - k * DIN;
                    av = hr[k] * xr[d];
                } else if (p < KT) {
                    av = xr[p - 32 * DIN];
                } else {
                    av = 0.f;
                }
                af[run * 8 + t] = (_Float16)av;
            }
        }
        v16h b0 = frag_from_runs(sBf + (c << 10) + (ncol << 5), grp);
        v16h b1 = frag_from_runs(sBf + (c << 10) + ((ncol + 16) << 5), grp);
        acc0 = __builtin_amdgcn_wmma_f32_16x16x32_f16(false, af, false, b0,
                                                      (short)0, acc0, false, false);
        acc1 = __builtin_amdgcn_wmma_f32_16x16x32_f16(false, af, false, b1,
                                                      (short)0, acc1, false, false);
    }

    // scatter-add into agg[dst] (C layout: vgpr r -> M=r(+8), N=lane&15)
    const int mhi = half * 8;
#pragma unroll
    for (int r = 0; r < 8; ++r) {
        int dn = sh_dst[wave][r + mhi];
        if (dn >= 0) {
            atomicAdd(&agg[(size_t)dn * 32 + ncol],      acc0[r]);
            atomicAdd(&agg[(size_t)dn * 32 + 16 + ncol], acc1[r]);
        }
    }
}

// ---------------------------------------------------------------- finalize convs
__global__ void __launch_bounds__(256) finalize_kernel(
    const float* __restrict__ x, const float* __restrict__ x2out,
    const float* __restrict__ agg11, const float* __restrict__ agg21,
    const float* __restrict__ cnt,
    const float* __restrict__ root11, const float* __restrict__ b11,
    const float* __restrict__ root21, const float* __restrict__ b21,
    float* __restrict__ out)
{
    int wave = threadIdx.x >> 5, c = threadIdx.x & 31;
    int n = blockIdx.x * 8 + wave;
    if (n >= NN) return;
    float s1 = 0.f, s2 = 0.f;
    for (int d = 0; d < 41; ++d) s1 += x[(size_t)n * 115 + d] * root11[d * 32 + c];
    for (int d = 0; d < 32; ++d) s2 += x2out[(size_t)n * 32 + d] * root21[d * 32 + c];
    float cc = fmaxf(cnt[n], 1.f);
    float x1f = fmaxf(s1 + agg11[(size_t)n * 32 + c] / cc + b11[c], 0.f);
    float x2f = fmaxf(s2 + agg21[(size_t)n * 32 + c] / cc + b21[c], 0.f);
    out[(size_t)n * 160 + c]      = x1f;
    out[(size_t)n * 160 + 32 + c] = x2f;
}

// ---------------------------------------------------------------- cross attention (1 wave/node)
__global__ void __launch_bounds__(256) attn_kernel(
    const float* __restrict__ qin, int qstride, int qoff, int dq,
    const float* __restrict__ kvin, int kvstride, int kvoff, int dkv,
    const float* __restrict__ qw, const float* __restrict__ qb,
    const float* __restrict__ kw, const float* __restrict__ kb,
    const float* __restrict__ vw, const float* __restrict__ vb,
    const float* __restrict__ ow, const float* __restrict__ ob,
    const float* __restrict__ gamma, const float* __restrict__ beta,
    float* __restrict__ out)
{
    int wave = threadIdx.x >> 5, c = threadIdx.x & 31;
    int n = blockIdx.x * 8 + wave;
    if (n >= NN) return;
    float Q = qb[c], K = kb[c], V = vb[c];
    for (int i = 0; i < dq; ++i)
        Q += qin[(size_t)n * qstride + qoff + i] * qw[i * 32 + c];
    for (int i = 0; i < dkv; ++i) {
        float xv = kvin[(size_t)n * kvstride + kvoff + i];
        K += xv * kw[i * 32 + c];
        V += xv * vw[i * 32 + c];
    }
    float mx = -3.4e38f;
    for (int j = 0; j < 32; ++j) mx = fmaxf(mx, Q * __shfl(K, j, 32));
    float sum = 0.f, o = 0.f;
    for (int j = 0; j < 32; ++j) {
        float ev = __expf(Q * __shfl(K, j, 32) - mx);
        sum += ev;
        o   += ev * __shfl(V, j, 32);
    }
    o /= sum;
    float y = ob[c];
    for (int j = 0; j < 32; ++j) y += __shfl(o, j, 32) * ow[j * 32 + c];
    y += Q;
    float mu = y;
    for (int off = 16; off; off >>= 1) mu += __shfl_xor(mu, off, 32);
    mu *= (1.f / 32.f);
    float dv = (y - mu) * (y - mu);
    for (int off = 16; off; off >>= 1) dv += __shfl_xor(dv, off, 32);
    dv *= (1.f / 32.f);
    out[(size_t)n * 32 + c] = (y - mu) * rsqrtf(dv + 1e-5f) * gamma[c] + beta[c];
}

// ---------------------------------------------------------------- trans 96x96 (WMMA)
__global__ void __launch_bounds__(256) trans_kernel(
    const float* __restrict__ glob, const float* __restrict__ grp,
    const float* __restrict__ atom,
    const float* __restrict__ w, const float* __restrict__ b,
    float* __restrict__ out)
{
    __shared__ __align__(16) _Float16 sWf[3 * 96 * 32];  // fragment order
    __shared__ float sb[96];
    const int tid = threadIdx.x, wave = tid >> 5, lane = tid & 31;
    for (int idx = tid; idx < 96 * 96; idx += 256) {
        int k = idx / 96, o = idx - (idx / 96) * 96;
        sWf[(((k >> 5) * 96 + o) << 5) + (k & 31)] = (_Float16)w[idx];
    }
    if (tid < 96) sb[tid] = b[tid];
    __syncthreads();

    const int m = lane & 15, ncol = lane & 15;
    const int grpk = (lane & 16) ? 8 : 0;
    const int node  = blockIdx.x * 128 + wave * 16 + m;
    const int nodec = node < NN ? node : NN - 1;  // clamp: loads unconditional

    v8f acc[6] = {};
    const float* srcs[3] = { glob, grp, atom };
#pragma unroll
    for (int c = 0; c < 3; ++c) {
        const float4* rp = (const float4*)(srcs[c] + (size_t)nodec * 32);
        float4 q0 = rp[grpk / 4],     q1 = rp[grpk / 4 + 1];
        float4 q2 = rp[4 + grpk / 4], q3 = rp[5 + grpk / 4];
        v16h af = __builtin_shufflevector(cvt8(q0, q1), cvt8(q2, q3),
                                          0, 1, 2, 3, 4, 5, 6, 7,
                                          8, 9, 10, 11, 12, 13, 14, 15);
#pragma unroll
        for (int nt = 0; nt < 6; ++nt) {
            v16h bf = frag_from_runs(sWf + ((c * 96 + nt * 16 + ncol) << 5), grpk);
            acc[nt] = __builtin_amdgcn_wmma_f32_16x16x32_f16(false, af, false, bf,
                                                             (short)0, acc[nt], false, false);
        }
    }
    const int mhi = (lane >> 4) * 8;
#pragma unroll
    for (int nt = 0; nt < 6; ++nt)
#pragma unroll
        for (int r = 0; r < 8; ++r) {
            int nd = blockIdx.x * 128 + wave * 16 + r + mhi;
            if (nd < NN) {
                float y = fmaxf(acc[nt][r] + sb[nt * 16 + ncol], 0.f);
                out[(size_t)nd * 160 + 64 + nt * 16 + ncol] = y;
            }
        }
}

// ---------------------------------------------------------------- launch
// pytree flatten order (sorted dict keys):
enum {
    IN_X = 0, IN_EI, IN_EA,
    A11_BIAS, A11_E1B, A11_E1W, A11_E2B, A11_E2W, A11_ROOT,
    A21_BIAS, A21_E1B, A21_E1W, A21_E2B, A21_E2W, A21_ROOT,
    AE_BETA, AE_GAMMA, AE_KB, AE_KW, AE_OB, AE_OW, AE_QB, AE_QW, AE_VB, AE_VW,
    GA_BETA, GA_GAMMA, GA_KB, GA_KW, GA_OB, GA_OW, GA_QB, GA_QW, GA_VB, GA_VW,
    GG_BETA, GG_GAMMA, GG_KB, GG_KW, GG_OB, GG_OW, GG_QB, GG_QW, GG_VB, GG_VW,
    AI_BETA, AI_GAMMA, AI_KB, AI_KW, AI_OB, AI_OW, AI_QB, AI_QW, AI_VB, AI_VW,
    CFC_FF1B, CFC_FF1W, CFC_FF2B, CFC_FF2W, CFC_TAB, CFC_TAW, CFC_TBB, CFC_TBW,
    TR_B, TR_W, X22_B, X22_W
};

extern "C" void kernel_launch(void* const* d_in, const int* in_sizes, int n_in,
                              void* d_out, int out_size, void* d_ws, size_t ws_size,
                              hipStream_t stream)
{
    #define FP(i) ((const float*)d_in[(i)])
    const float* x      = FP(IN_X);
    const int*   ei     = (const int*)d_in[IN_EI];
    const float* eattr  = FP(IN_EA);
    float* ws  = (float*)d_ws;
    float* out = (float*)d_out;

    float* x2out = ws + WS_X2OUT;
    float* agg21 = ws + WS_AGG21;
    float* agg11 = ws + WS_AGG11;
    float* cnt   = ws + WS_CNT;
    float* inter = ws + WS_INTER;
    float* glob  = ws + WS_GLOB;
    float* grp   = ws + WS_GRP;
    float* atom  = ws + WS_ATOM;

    // zero accumulators (agg21|agg11|cnt contiguous)
    hipMemsetAsync((void*)agg21, 0, (size_t)(1280000 * 2 + 40000) * sizeof(float), stream);

    cfc_kernel<<<(NN + 255) / 256, 256, 0, stream>>>(
        x, FP(CFC_FF1W), FP(CFC_FF1B), FP(CFC_FF2W), FP(CFC_FF2B),
        FP(CFC_TAW), FP(CFC_TAB), FP(CFC_TBW), FP(CFC_TBB),
        FP(X22_W), FP(X22_B), x2out);

    count_kernel<<<(NE + 255) / 256, 256, 0, stream>>>(ei, cnt);

    // a21: xn = x2out (stride 32, din 32); a11: xn = x1 = x[:, :41] (stride 115, din 41)
    nnconv_kernel<32><<<(NE + 127) / 128, 256, 0, stream>>>(
        x2out, 32, FP(A21_E1W), FP(A21_E1B), FP(A21_E2W), FP(A21_E2B),
        ei, eattr, agg21);
    nnconv_kernel<41><<<(NE + 127) / 128, 256, 0, stream>>>(
        x, 115, FP(A11_E1W), FP(A11_E1B), FP(A11_E2W), FP(A11_E2B),
        ei, eattr, agg11);

    finalize_kernel<<<NN / 8, 256, 0, stream>>>(
        x, x2out, agg11, agg21, cnt,
        FP(A11_ROOT), FP(A11_BIAS), FP(A21_ROOT), FP(A21_BIAS), out);

    // inter = attn(attn_inter, g = x[:,73:94], G = x[:,94:115])
    attn_kernel<<<NN / 8, 256, 0, stream>>>(
        x, 115, 73, 21,  x, 115, 94, 21,
        FP(AI_QW), FP(AI_QB), FP(AI_KW), FP(AI_KB), FP(AI_VW), FP(AI_VB),
        FP(AI_OW), FP(AI_OB), FP(AI_GAMMA), FP(AI_BETA), inter);
    // glob = attn(attn_gg, x3 = x[:,47:68], inter)
    attn_kernel<<<NN / 8, 256, 0, stream>>>(
        x, 115, 47, 21,  inter, 32, 0, 32,
        FP(GG_QW), FP(GG_QB), FP(GG_KW), FP(GG_KB), FP(GG_VW), FP(GG_VB),
        FP(GG_OW), FP(GG_OB), FP(GG_GAMMA), FP(GG_BETA), glob);
    // grp = attn(attn_ga, x1f (out[:,0:32]), glob)
    attn_kernel<<<NN / 8, 256, 0, stream>>>(
        out, 160, 0, 32,  glob, 32, 0, 32,
        FP(GA_QW), FP(GA_QB), FP(GA_KW), FP(GA_KB), FP(GA_VW), FP(GA_VB),
        FP(GA_OW), FP(GA_OB), FP(GA_GAMMA), FP(GA_BETA), grp);
    // atom = attn(attn_ae, x2f (out[:,32:64]), grp)
    attn_kernel<<<NN / 8, 256, 0, stream>>>(
        out, 160, 32, 32,  grp, 32, 0, 32,
        FP(AE_QW), FP(AE_QB), FP(AE_KW), FP(AE_KB), FP(AE_VW), FP(AE_VB),
        FP(AE_OW), FP(AE_OB), FP(AE_GAMMA), FP(AE_BETA), atom);

    trans_kernel<<<(NN + 127) / 128, 256, 0, stream>>>(
        glob, grp, atom, FP(TR_W), FP(TR_B), out);
    #undef FP
}
